// LongformerBlock_69020124446891
// MI455X (gfx1250) — matrix operand order repaired and checked by reading, not compiled, not run on hardware
//
#include <hip/hip_runtime.h>
#include <hip/hip_bf16.h>

typedef __attribute__((ext_vector_type(16))) _Float16 v16h;
typedef __attribute__((ext_vector_type(8)))  _Float16 v8h;
typedef __attribute__((ext_vector_type(8)))  float    v8f;

static constexpr int B_  = 2;
static constexpr int S_  = 2048;
static constexpr int D_  = 512;
static constexpr int H_  = 8;
static constexpr int HD_ = 64;
static constexpr int MLP_ = 2048;
static constexpr int T_  = B_ * S_;      // 4096 tokens
static constexpr int HW_ = 256;          // WINDOW/2
static constexpr int NG_ = 64;           // global tokens
#define NEGINF_SCORE (-1.0e10f)

// ---------------------------------------------------------------------------
// WMMA helpers (CDNA5 f16 16x16x32, f32 accumulate)
// ---------------------------------------------------------------------------
__device__ __forceinline__ v8f wmma_f16(v16h a, v16h b, v8f c) {
  return __builtin_amdgcn_wmma_f32_16x16x32_f16(false, a, false, b,
                                                (short)0, c, false, false);
}

union FragU { v16h v; v8h h[2]; };

// A-matrix fragment (16x32 f16): lane (m = lane&15, hh = lane>>4) holds
// K = [8*hh .. 8*hh+7] and [16+8*hh .. 16+8*hh+7]  -> two contiguous 16B runs.
__device__ __forceinline__ v16h load_a_frag(const _Float16* A, int lda,
                                            int m0, int k0, int lane) {
  int m = lane & 15, hh = lane >> 4;
  const _Float16* base = A + (size_t)(m0 + m) * lda + k0 + 8 * hh;
  FragU u;
  u.h[0] = *(const v8h*)(base);
  u.h[1] = *(const v8h*)(base + 16);
  return u.v;
}

// B-matrix fragment (32x16 f16) loaded from a K-contiguous ("transposed")
// matrix Bt where Bt[n][k] = B[k][n], row stride ldb:
// lane (n = lane&15, hh = lane>>4) holds K = [8*hh .. 8*hh+7] and [16+8*hh .. 16+8*hh+7].
__device__ __forceinline__ v16h load_bt_frag(const _Float16* Bt, int ldb,
                                             int k0, int n0, int lane) {
  int n = lane & 15, hh = lane >> 4;
  const _Float16* base = Bt + (size_t)(n0 + n) * ldb + k0 + 8 * hh;
  FragU u;
  u.h[0] = *(const v8h*)(base);
  u.h[1] = *(const v8h*)(base + 16);
  return u.v;
}

// ---------------------------------------------------------------------------
// Weight convert + transpose: W[K][N] f32  ->  Wt[N][K] f16 (optional scale)
// ---------------------------------------------------------------------------
__global__ void convT_kernel(const float* __restrict__ W, _Float16* __restrict__ Wt,
                             int K, int N, float scale) {
  int idx = blockIdx.x * 256 + threadIdx.x;
  if (idx >= K * N) return;
  int k = idx / N, n = idx % N;
  Wt[(size_t)n * K + k] = (_Float16)(W[idx] * scale);
}

// ---------------------------------------------------------------------------
// LayerNorm over D=512, one wave per row, f16 output
// ---------------------------------------------------------------------------
__global__ __launch_bounds__(256) void ln_kernel(const float* __restrict__ X,
                                                 const float* __restrict__ scale,
                                                 const float* __restrict__ bias,
                                                 _Float16* __restrict__ Y,
                                                 int rows) {
  int wave = threadIdx.x >> 5, lane = threadIdx.x & 31;
  int row = blockIdx.x * 8 + wave;
  if (row >= rows) return;
  const float* x = X + (size_t)row * D_;
  float v[16];
  float s = 0.0f;
#pragma unroll
  for (int i = 0; i < 16; ++i) { v[i] = x[lane + 32 * i]; s += v[i]; }
#pragma unroll
  for (int m = 1; m < 32; m <<= 1) s += __shfl_xor(s, m, 32);
  float mean = s * (1.0f / D_);
  float ss = 0.0f;
#pragma unroll
  for (int i = 0; i < 16; ++i) { float d = v[i] - mean; ss += d * d; }
#pragma unroll
  for (int m = 1; m < 32; m <<= 1) ss += __shfl_xor(ss, m, 32);
  float r = rsqrtf(ss * (1.0f / D_) + 1e-6f);
  _Float16* y = Y + (size_t)row * D_;
#pragma unroll
  for (int i = 0; i < 16; ++i) {
    int c = lane + 32 * i;
    y[c] = (_Float16)((v[i] - mean) * r * scale[c] + bias[c]);
  }
}

// ---------------------------------------------------------------------------
// Generic WMMA GEMM, 4x4 register-blocked: one wave owns a 64x64 C tile.
// Per 32-deep k-step: 4 A-frags + 4 B-frags -> 16 WMMAs (4x data reuse).
// C[M][N] = A[M][K](f16,row-major) x Bt[N][K](f16)
// MODE 1: C -> f16 row-major
// MODE 2: C -> f16 scattered as Vt[b][h][hd][s]
// MODE 3: C + resid -> f32 row-major
// MODE 4: gelu(C + bias) -> f16 row-major
// MODE 5: C + bias + resid -> f32 row-major
// ---------------------------------------------------------------------------
template <int MODE>
__global__ __launch_bounds__(256) void gemm_kernel(const _Float16* __restrict__ A,
                                                   const _Float16* __restrict__ Bt,
                                                   int M, int N, int K,
                                                   _Float16* __restrict__ Cf16,
                                                   float* __restrict__ Cf32,
                                                   const float* __restrict__ bias,
                                                   const float* __restrict__ resid) {
  int wave = threadIdx.x >> 5, lane = threadIdx.x & 31;
  int n64 = N >> 6;
  int tile = blockIdx.x * 8 + wave;
  if (tile >= (M >> 6) * n64) return;
  int m0 = (tile / n64) << 6;
  int n0 = (tile % n64) << 6;

  v8f acc[4][4] = {};
  for (int k0 = 0; k0 < K; k0 += 32) {
    v16h af[4], bf[4];
#pragma unroll
    for (int i = 0; i < 4; ++i) af[i] = load_a_frag(A, K, m0 + 16 * i, k0, lane);
#pragma unroll
    for (int j = 0; j < 4; ++j) bf[j] = load_bt_frag(Bt, K, k0, n0 + 16 * j, lane);
    if (k0 + 32 < K) {  // hint next k-step into cache (global_prefetch_b8)
      __builtin_prefetch(A + (size_t)(m0 + (lane & 15)) * K + k0 + 32, 0, 3);
      __builtin_prefetch(Bt + (size_t)(n0 + (lane & 15)) * K + k0 + 32, 0, 3);
    }
#pragma unroll
    for (int i = 0; i < 4; ++i)
#pragma unroll
      for (int j = 0; j < 4; ++j)
        acc[i][j] = wmma_f16(af[i], bf[j], acc[i][j]);
  }

  int half = lane >> 4, nn = lane & 15;
#pragma unroll
  for (int i = 0; i < 4; ++i) {
#pragma unroll
    for (int j = 0; j < 4; ++j) {
      int col = n0 + 16 * j + nn;
#pragma unroll
      for (int r = 0; r < 8; ++r) {
        int row = m0 + 16 * i + r + 8 * half;
        float v = acc[i][j][r];
        if (MODE == 1) {
          Cf16[(size_t)row * N + col] = (_Float16)v;
        } else if (MODE == 2) {
          int bb = row >> 11, s = row & (S_ - 1);
          int hh = col >> 6, hd = col & (HD_ - 1);
          Cf16[(((size_t)(bb * H_ + hh)) * HD_ + hd) * S_ + s] = (_Float16)v;
        } else if (MODE == 3) {
          Cf32[(size_t)row * N + col] = v + resid[(size_t)row * N + col];
        } else if (MODE == 4) {
          float x = v + bias[col];
          float g = 0.5f * x * (1.0f + tanhf(0.7978845608f * (x + 0.044715f * x * x * x)));
          Cf16[(size_t)row * N + col] = (_Float16)g;
        } else if (MODE == 5) {
          Cf32[(size_t)row * N + col] = v + bias[col] + resid[(size_t)row * N + col];
        }
      }
    }
  }
}

// ---------------------------------------------------------------------------
// Longformer attention, flash style. 1 wave per (b, h, 16-query tile).
// Q/K f16 [b*S][H*HD] row-major; V f16 transposed Vt[b][h][hd][s].
// ---------------------------------------------------------------------------
__global__ __launch_bounds__(256) void attn_kernel(const _Float16* __restrict__ Qf,
                                                   const _Float16* __restrict__ Kf,
                                                   const _Float16* __restrict__ Vt,
                                                   _Float16* __restrict__ Ctx) {
  __shared__ _Float16 lds[8 * 16 * 32];   // per-wave 16x32 P tile
  int wave = threadIdx.x >> 5, lane = threadIdx.x & 31;
  int gw = blockIdx.x * 8 + wave;                 // 0..2047
  int b = gw >> 10;                               // H*(S/16)=1024 tiles/batch
  int rem = gw & 1023;
  int h = rem >> 7;
  int q0 = (rem & 127) << 4;

  const _Float16* Qb = Qf + (size_t)b * S_ * (H_ * HD_) + h * HD_;
  const _Float16* Kb = Kf + (size_t)b * S_ * (H_ * HD_) + h * HD_;
  const _Float16* Vb = Vt + ((size_t)(b * H_ + h) * HD_) * S_;
  _Float16* myl = &lds[wave * 512];

  int half = lane >> 4, nn = lane & 15;

  v16h qa0 = load_a_frag(Qb, H_ * HD_, q0, 0, lane);
  v16h qa1 = load_a_frag(Qb, H_ * HD_, q0, 32, lane);

  v8f c0 = {}, c1 = {}, c2 = {}, c3 = {};
  float mrow[8], lrow[8];
#pragma unroll
  for (int r = 0; r < 8; ++r) { mrow[r] = -3.0e38f; lrow[r] = 0.0f; }

  auto process = [&](int kb) {   // 32 keys starting at kb
    // ---- scores: two 16x16 tiles, K-dim = HD = 64 (two wmma steps each) ----
    v8f s0 = {}, s1 = {};
    s0 = wmma_f16(qa0, load_bt_frag(Kb, H_ * HD_, 0,  kb,      lane), s0);
    s0 = wmma_f16(qa1, load_bt_frag(Kb, H_ * HD_, 32, kb,      lane), s0);
    s1 = wmma_f16(qa0, load_bt_frag(Kb, H_ * HD_, 0,  kb + 16, lane), s1);
    s1 = wmma_f16(qa1, load_bt_frag(Kb, H_ * HD_, 32, kb + 16, lane), s1);

    // ---- mask: band |i-j|<=256  OR  j<64  OR  i<64 ----
#pragma unroll
    for (int r = 0; r < 8; ++r) {
      int i = q0 + r + 8 * half;
      int j0 = kb + nn, j1 = kb + 16 + nn;
      int d0 = i > j0 ? i - j0 : j0 - i;
      int d1 = i > j1 ? i - j1 : j1 - i;
      bool g = (i < NG_);
      if (!((d0 <= HW_) || (j0 < NG_) || g)) s0[r] = NEGINF_SCORE;
      if (!((d1 <= HW_) || (j1 < NG_) || g)) s1[r] = NEGINF_SCORE;
    }

    // ---- online softmax (rows live in 16-lane halves, xor 1..8) ----
    float p0[8], p1[8];
#pragma unroll
    for (int r = 0; r < 8; ++r) {
      float t = fmaxf(s0[r], s1[r]);
#pragma unroll
      for (int m = 1; m <= 8; m <<= 1) t = fmaxf(t, __shfl_xor(t, m, 32));
      float mn = fmaxf(mrow[r], t);
      float sc = __expf(mrow[r] - mn);
      mrow[r] = mn;
      p0[r] = __expf(s0[r] - mn);
      p1[r] = __expf(s1[r] - mn);
      float rs = p0[r] + p1[r];
#pragma unroll
      for (int m = 1; m <= 8; m <<= 1) rs += __shfl_xor(rs, m, 32);
      lrow[r] = lrow[r] * sc + rs;
      c0[r] *= sc; c1[r] *= sc; c2[r] *= sc; c3[r] *= sc;
    }

    // ---- C-layout -> A-layout transpose of P through per-wave LDS ----
#pragma unroll
    for (int r = 0; r < 8; ++r) {
      int row = r + 8 * half;
      myl[row * 32 + nn]      = (_Float16)p0[r];
      myl[row * 32 + 16 + nn] = (_Float16)p1[r];
    }
    v16h pa = load_a_frag(myl, 32, 0, 0, lane);   // 16x32 A fragment

    // ---- ctx += P @ V, four 16-wide HD chunks ----
    c0 = wmma_f16(pa, load_bt_frag(Vb, S_, kb, 0,  lane), c0);
    c1 = wmma_f16(pa, load_bt_frag(Vb, S_, kb, 16, lane), c1);
    c2 = wmma_f16(pa, load_bt_frag(Vb, S_, kb, 32, lane), c2);
    c3 = wmma_f16(pa, load_bt_frag(Vb, S_, kb, 48, lane), c3);
  };

  if (q0 < NG_) {
    for (int kb = 0; kb < S_; kb += 32) process(kb);
  } else {
    process(0); process(32);                       // global columns [0,64)
    int lo = q0 - HW_; if (lo < NG_) lo = NG_;
    lo &= ~31;
    int hi = q0 + 15 + HW_; if (hi > S_ - 1) hi = S_ - 1;
    for (int kb = lo; kb <= hi; kb += 32) process(kb);
  }

  // ---- normalize + store ctx (f16 row-major [token][h*64+hd]) ----
#pragma unroll
  for (int r = 0; r < 8; ++r) {
    float inv = 1.0f / lrow[r];
    int row = q0 + r + 8 * half;
    size_t base = ((size_t)b * S_ + row) * (H_ * HD_) + h * HD_;
    Ctx[base + 0  + nn] = (_Float16)(c0[r] * inv);
    Ctx[base + 16 + nn] = (_Float16)(c1[r] * inv);
    Ctx[base + 32 + nn] = (_Float16)(c2[r] * inv);
    Ctx[base + 48 + nn] = (_Float16)(c3[r] * inv);
  }
}

// ---------------------------------------------------------------------------
// Host-side launcher
// ---------------------------------------------------------------------------
extern "C" void kernel_launch(void* const* d_in, const int* in_sizes, int n_in,
                              void* d_out, int out_size, void* d_ws, size_t ws_size,
                              hipStream_t stream) {
  (void)in_sizes; (void)n_in; (void)out_size; (void)ws_size;

  const float* inputs = (const float*)d_in[0];
  // d_in[1] = global_mask (statically first NG_ tokens; not needed)
  const float* ln1_s = (const float*)d_in[2];
  const float* ln1_b = (const float*)d_in[3];
  const float* wq    = (const float*)d_in[4];
  const float* wk    = (const float*)d_in[5];
  const float* wv    = (const float*)d_in[6];
  const float* wo    = (const float*)d_in[7];
  const float* ln2_s = (const float*)d_in[8];
  const float* ln2_b = (const float*)d_in[9];
  const float* w1    = (const float*)d_in[10];
  const float* b1    = (const float*)d_in[11];
  const float* w2    = (const float*)d_in[12];
  const float* b2    = (const float*)d_in[13];
  float* out = (float*)d_out;

  // workspace layout (256B aligned slabs)
  char* ws = (char*)d_ws;
  size_t off = 0;
  auto alloc = [&](size_t bytes) { void* p = ws + off; off += (bytes + 255) & ~(size_t)255; return p; };
  _Float16* xln1 = (_Float16*)alloc((size_t)T_ * D_ * 2);
  _Float16* wqT  = (_Float16*)alloc((size_t)D_ * D_ * 2);
  _Float16* wkT  = (_Float16*)alloc((size_t)D_ * D_ * 2);
  _Float16* wvT  = (_Float16*)alloc((size_t)D_ * D_ * 2);
  _Float16* woT  = (_Float16*)alloc((size_t)D_ * D_ * 2);
  _Float16* w1T  = (_Float16*)alloc((size_t)D_ * MLP_ * 2);
  _Float16* w2T  = (_Float16*)alloc((size_t)MLP_ * D_ * 2);
  _Float16* Qf   = (_Float16*)alloc((size_t)T_ * D_ * 2);
  _Float16* Kf   = (_Float16*)alloc((size_t)T_ * D_ * 2);
  _Float16* Vtf  = (_Float16*)alloc((size_t)T_ * D_ * 2);
  _Float16* ctx  = (_Float16*)alloc((size_t)T_ * D_ * 2);
  float*    x32  = (float*)   alloc((size_t)T_ * D_ * 4);
  _Float16* xln2 = (_Float16*)alloc((size_t)T_ * D_ * 2);
  _Float16* hact = (_Float16*)alloc((size_t)T_ * MLP_ * 2);

  // 1) weight convert/transpose to f16 (Q scale 1/sqrt(HD) baked in)
  convT_kernel<<<(D_ * D_) / 256, 256, 0, stream>>>(wq, wqT, D_, D_, 0.125f);
  convT_kernel<<<(D_ * D_) / 256, 256, 0, stream>>>(wk, wkT, D_, D_, 1.0f);
  convT_kernel<<<(D_ * D_) / 256, 256, 0, stream>>>(wv, wvT, D_, D_, 1.0f);
  convT_kernel<<<(D_ * D_) / 256, 256, 0, stream>>>(wo, woT, D_, D_, 1.0f);
  convT_kernel<<<(D_ * MLP_) / 256, 256, 0, stream>>>(w1, w1T, D_, MLP_, 1.0f);
  convT_kernel<<<(MLP_ * D_) / 256, 256, 0, stream>>>(w2, w2T, MLP_, D_, 1.0f);

  // 2) LN1
  ln_kernel<<<T_ / 8, 256, 0, stream>>>(inputs, ln1_s, ln1_b, xln1, T_);

  // 3) QKV projections (V stored transposed per head)
  int waves512 = (T_ / 64) * (D_ / 64);           // 512 waves -> 64 blocks
  gemm_kernel<1><<<waves512 / 8, 256, 0, stream>>>(xln1, wqT, T_, D_, D_, Qf, nullptr, nullptr, nullptr);
  gemm_kernel<1><<<waves512 / 8, 256, 0, stream>>>(xln1, wkT, T_, D_, D_, Kf, nullptr, nullptr, nullptr);
  gemm_kernel<2><<<waves512 / 8, 256, 0, stream>>>(xln1, wvT, T_, D_, D_, Vtf, nullptr, nullptr, nullptr);

  // 4) Longformer attention: 2048 waves -> 256 blocks
  attn_kernel<<<256, 256, 0, stream>>>(Qf, Kf, Vtf, ctx);

  // 5) output proj + residual -> x (f32)
  gemm_kernel<3><<<waves512 / 8, 256, 0, stream>>>(ctx, woT, T_, D_, D_, nullptr, x32, nullptr, inputs);

  // 6) LN2
  ln_kernel<<<T_ / 8, 256, 0, stream>>>(x32, ln2_s, ln2_b, xln2, T_);

  // 7) MLP up + GELU
  int wavesM1 = (T_ / 64) * (MLP_ / 64);          // 2048 waves -> 256 blocks
  gemm_kernel<4><<<wavesM1 / 8, 256, 0, stream>>>(xln2, w1T, T_, MLP_, D_, hact, nullptr, b1, nullptr);

  // 8) MLP down + bias + residual -> out (f32)
  gemm_kernel<5><<<waves512 / 8, 256, 0, stream>>>(hact, w2T, T_, D_, MLP_, nullptr, out, b2, x32);
}
